// EquivariantBlock_38431367365236
// MI455X (gfx1250) — matrix operand order repaired and checked by reading, not compiled, not run on hardware
//
#include <hip/hip_runtime.h>
#include <hip/hip_bf16.h>

// ---------------------------------------------------------------------------
// Problem constants (from the reference)
// ---------------------------------------------------------------------------
#define NN    16384      // nodes
#define EE    65536      // edges
#define HH    256        // hidden
#define AHD   128        // attention hidden
#define NHEAD 4
#define HDIM  32         // AHD / NHEAD
#define DWK   258        // H + 2  (w feature dim)
#define CRANGE 10.0f

typedef __attribute__((ext_vector_type(16))) __bf16 v16bf;
typedef __attribute__((ext_vector_type(8)))  __bf16 v8bf;
typedef __attribute__((ext_vector_type(8)))  float  v8f;
typedef __attribute__((ext_vector_type(4)))  float  v4f;

__device__ __forceinline__ float siluf(float x) { return x / (1.0f + __expf(-x)); }

// WMMA D = A(16x32 bf16) * B(32x16 bf16) + C(16x16 f32)
__device__ __forceinline__ v8f wmma_bf16(v16bf a, v16bf b, v8f c) {
    return __builtin_amdgcn_wmma_f32_16x16x32_bf16(false, a, false, b, (short)0, c,
                                                   false, false);
}

// Load an A fragment (16x32, M x K) from an LDS bf16 tile, row-major, stride in
// elements.  Lanes 0-15 hold rows m=0..15 with k = {kt*32+0..7, kt*32+16..23};
// lanes 16-31 hold k = {+8..15, +24..31}.  Two contiguous 16B chunks ->
// two ds_load_b128.
__device__ __forceinline__ v16bf load_a_frag(const __bf16* base, int stride,
                                             int mrow0, int lane, int kt) {
    int row = mrow0 + (lane & 15);
    int k0  = kt * 32 + ((lane & 16) ? 8 : 0);
    const __bf16* p = base + row * stride + k0;
    union { v16bf v; v8bf h[2]; } u;
    u.h[0] = *(const v8bf*)(p);
    u.h[1] = *(const v8bf*)(p + 16);
    return u.v;
}

// 32-row (2 M-tile) x NT-n-tile GEMM strip.  Each B fragment feeds 2 WMMAs and
// is double-buffered in registers (the next fragment's loads are in flight
// while the current one is consumed).  Register budget: acc 2*NT*8 + A 16 +
// B 16  ->  <=128 VGPRs for NT=4, no accumulator spills.
template<int KT, int NT>
__device__ __forceinline__ void gemm_strip2(const __bf16* Atile, int astride,
                                            int mrow0, const __bf16* Bf,
                                            int btilesN, int ntile0, int lane,
                                            v8f acc[2][NT]) {
    const __bf16* bbase = Bf + (size_t)lane * 16;
#define LOADB(kt, nt) (*(const v16bf*)(bbase + (size_t)((kt) * btilesN + ntile0 + (nt)) * 512))
    v16bf bc = LOADB(0, 0);
#pragma unroll 1
    for (int kt = 0; kt < KT; ++kt) {
        v16bf a0 = load_a_frag(Atile, astride, mrow0,      lane, kt);
        v16bf a1 = load_a_frag(Atile, astride, mrow0 + 16, lane, kt);
#pragma unroll
        for (int nt = 0; nt < NT; ++nt) {
            // issue next B load before consuming the current one
            int nkt = (nt + 1 < NT) ? kt : ((kt + 1 < KT) ? kt + 1 : 0);
            int nnt = (nt + 1 < NT) ? nt + 1 : 0;
            v16bf bn = LOADB(nkt, nnt);
            acc[0][nt] = wmma_bf16(a0, bc, acc[0][nt]);
            acc[1][nt] = wmma_bf16(a1, bc, acc[1][nt]);
            bc = bn;
        }
    }
#undef LOADB
}

// ---------------------------------------------------------------------------
// Workspace layout (byte offsets)
// ---------------------------------------------------------------------------
#define OFF_NODE_EQ  0u                         // f32 [4][NN][3]
#define OFF_CNT0     786432u                    // f32 [4][NN]
#define OFF_CNT1     1048576u
#define OFF_SUM0     1310720u
#define OFF_SUM1     1572864u
#define OFF_W1FRAG   1835008u                   // bf16 4 * (16*32) frags * 512
#define OFF_W1LAST   3932160u                   // f32 [4][512]
#define OFF_W2FRAG   3940352u                   // bf16 4 * (16*16) frags * 512
#define OFF_W3FRAG   4988928u                   // bf16 4 * (8*8)  frags * 512
#define OFF_WQFRAG   5251072u                   // bf16 (9*8) frags * 512
#define OFF_WKFRAG   5324800u
#define OFF_WVFRAG   5398528u
#define ZERO_FLOATS  458752                     // node_eq + cnt/sum accumulators

// ---------------------------------------------------------------------------
// Utility kernels
// ---------------------------------------------------------------------------
__global__ void k_zero(float* p, int n) {
    int i = blockIdx.x * 256 + threadIdx.x;
    if (i < n) p[i] = 0.0f;
}

// Convert row-major f32 weight [srcK x Nn] (batched) into WMMA-B fragment-major
// bf16: frag (kt,nt) -> 32 lanes x 16 contiguous bf16.  Rows >= srcK -> 0.
__global__ void k_fragconv(const float* __restrict__ src, __bf16* __restrict__ dst,
                           int srcK, int Nn, int Kt, int Ntiles, int batch,
                           long srcBatchStride) {
    long gid = (long)blockIdx.x * 256 + threadIdx.x;
    long total = (long)batch * Kt * Ntiles * 32;
    if (gid >= total) return;
    int  lane = (int)(gid & 31);
    long f    = gid >> 5;
    int  nt   = (int)(f % Ntiles);  long f2 = f / Ntiles;
    int  kt   = (int)(f2 % Kt);     int  b  = (int)(f2 / Kt);
    int  n     = nt * 16 + (lane & 15);
    int  kbase = kt * 32 + ((lane & 16) ? 8 : 0);
    __bf16* o  = dst + gid * 16;
#pragma unroll
    for (int e = 0; e < 16; ++e) {
        int k = kbase + (e < 8 ? e : e + 8);
        float v = (k < srcK) ? src[b * srcBatchStride + (long)k * Nn + n] : 0.0f;
        o[e] = (__bf16)v;
    }
}

// Extract the 513th row of W1 (the edge_attr column weights) as f32.
__global__ void k_w1last(const float* __restrict__ W1, float* __restrict__ out) {
    int i = blockIdx.x * 256 + threadIdx.x;      // 4*512
    if (i >= 4 * 512) return;
    int t = i >> 9, n = i & 511;
    out[i] = W1[(size_t)t * 513 * 512 + (size_t)512 * 512 + n];
}

// Per-node masked edge counts and edge_attr sums (for mean0/mean1).
__global__ void k_edge_stats(const int* __restrict__ eidx,
                             const unsigned char* __restrict__ emask,
                             const float* __restrict__ eattr,
                             float* __restrict__ cnt0, float* __restrict__ cnt1,
                             float* __restrict__ sum0, float* __restrict__ sum1) {
    int e = blockIdx.x * 256 + threadIdx.x;
    if (e >= EE) return;
    int row = eidx[e], col = eidx[EE + e];
    float ea = eattr[e];
#pragma unroll
    for (int t = 0; t < 4; ++t) {
        if (emask[(size_t)t * EE + e]) {
            atomicAdd(cnt0 + t * NN + row, 1.0f);
            atomicAdd(cnt1 + t * NN + col, 1.0f);
            atomicAdd(sum0 + t * NN + row, ea);
            atomicAdd(sum1 + t * NN + col, ea);
        }
    }
}

// ---------------------------------------------------------------------------
// Fused edge MLP: 64 edges x 1 type per block, 8 waves.
// Each wave-pass computes a 32-row x 64-col strip (acc[2][4]).
// LDS: bufA (h_pair then X2) | bufB (X1 then X3) | ea/rows/cols/partials
// ---------------------------------------------------------------------------
#define STR1 520   // stride for K=512 tiles (bank-conflict-free ds_load_b128)
#define STR2 264   // K=256
#define STR3 136   // K=128
#define SM_A     0u
#define SM_B     (64u * STR1 * 2u)             // 66560
#define SM_EA    (SM_B + 64u * STR1 * 2u)      // 133120
#define SM_ROW   (SM_EA + 256u)
#define SM_COL   (SM_ROW + 256u)
#define SM_PART  (SM_COL + 256u)
#define SM_EDGE_TOTAL (SM_PART + 1024u)        // 134912

__global__ void __launch_bounds__(256, 1)
k_edge_mlp(const float* __restrict__ h,
           const float* __restrict__ coord,
           const float* __restrict__ eattr,
           const float* __restrict__ elen,
           const int* __restrict__ eidx,
           const unsigned char* __restrict__ emask,
           const __bf16* __restrict__ w1f,
           const float* __restrict__ w1last,
           const float* __restrict__ b1,
           const __bf16* __restrict__ w2f,
           const float* __restrict__ b2,
           const __bf16* __restrict__ w3f,
           const float* __restrict__ b3,
           const float* __restrict__ W4,
           const float* __restrict__ b4,
           float* __restrict__ node_eq) {
    extern __shared__ char smem[];
    __bf16* bufA  = (__bf16*)(smem + SM_A);
    __bf16* bufB  = (__bf16*)(smem + SM_B);
    float*  ea_s  = (float*)(smem + SM_EA);
    int*    row_s = (int*)(smem + SM_ROW);
    int*    col_s = (int*)(smem + SM_COL);
    float*  part  = (float*)(smem + SM_PART);

    const int tid  = threadIdx.x;
    const int lane = tid & 31;
    const int wv   = tid >> 5;            // 0..7
    const int t    = blockIdx.y;          // type
    const int e0   = blockIdx.x * 64;
    const int mlo  = (lane & 16) ? 8 : 0;
    const int nlan = lane & 15;

    if (tid < 64) {
        int e = e0 + tid;
        row_s[tid] = eidx[e];
        col_s[tid] = eidx[EE + e];
        ea_s[tid]  = eattr[e];
    }
    __syncthreads();

    // ---- gather h_pair tile (64 x 512) into bufA as bf16 -------------------
    for (int it = tid; it < 64 * 128; it += 256) {
        int r  = it >> 7;
        int c  = (it & 127) * 4;
        int nd = (c < 256) ? row_s[r] : col_s[r];
        int cc = (c < 256) ? c : c - 256;
        v4f f  = *(const v4f*)(h + ((size_t)nd * 4 + t) * HH + cc);
        __bf16* q = bufA + r * STR1 + c;
        q[0] = (__bf16)f.x; q[1] = (__bf16)f.y; q[2] = (__bf16)f.z; q[3] = (__bf16)f.w;
    }
    __syncthreads();

    // ---- stage 1: X1 = silu(h_pair @ W1 + ea*w1last + b1)   [64x512] -------
    // 128 output tiles = 16 strip-units of (2 M-tiles x 4 n-tiles); 2 passes.
    {
        const __bf16* Bf = w1f + (size_t)t * (16 * 32) * 512;
#pragma unroll 1
        for (int p = 0; p < 2; ++p) {
            int unit   = wv + p * 8;       // 0..15
            int mpair  = unit & 1;
            int ntile0 = (unit >> 1) * 4;
            v8f acc[2][4];
#pragma unroll
            for (int i = 0; i < 2; ++i)
#pragma unroll
                for (int j = 0; j < 4; ++j) acc[i][j] = (v8f){0,0,0,0,0,0,0,0};
            gemm_strip2<16, 4>(bufA, STR1, mpair * 32, Bf, 32, ntile0, lane, acc);
#pragma unroll
            for (int mi = 0; mi < 2; ++mi) {
                int mrow0 = mpair * 32 + mi * 16;
                float eav[8];
#pragma unroll
                for (int r = 0; r < 8; ++r) eav[r] = ea_s[mrow0 + r + mlo];
#pragma unroll
                for (int nt = 0; nt < 4; ++nt) {
                    int n  = (ntile0 + nt) * 16 + nlan;
                    float wl = w1last[t * 512 + n];
                    float bb = b1[t * 512 + n];
#pragma unroll
                    for (int r = 0; r < 8; ++r) {
                        float x = acc[mi][nt][r] + eav[r] * wl + bb;
                        bufB[(mrow0 + r + mlo) * STR1 + n] = (__bf16)siluf(x);
                    }
                }
            }
        }
    }
    __syncthreads();

    // ---- stage 2: X2 = silu(X1 @ W2 + b2)   [64x256] -----------------------
    // 64 output tiles = 8 strip-units of (2 M-tiles x 4 n-tiles); 1 pass.
    {
        const __bf16* Bf = w2f + (size_t)t * (16 * 16) * 512;
        int mpair  = wv & 1;
        int ntile0 = (wv >> 1) * 4;
        v8f acc[2][4];
#pragma unroll
        for (int i = 0; i < 2; ++i)
#pragma unroll
            for (int j = 0; j < 4; ++j) acc[i][j] = (v8f){0,0,0,0,0,0,0,0};
        gemm_strip2<16, 4>(bufB, STR1, mpair * 32, Bf, 16, ntile0, lane, acc);
#pragma unroll
        for (int mi = 0; mi < 2; ++mi) {
            int mrow0 = mpair * 32 + mi * 16;
#pragma unroll
            for (int nt = 0; nt < 4; ++nt) {
                int n = (ntile0 + nt) * 16 + nlan;
                float bb = b2[t * 256 + n];
#pragma unroll
                for (int r = 0; r < 8; ++r) {
                    float x = acc[mi][nt][r] + bb;
                    bufA[(mrow0 + r + mlo) * STR2 + n] = (__bf16)siluf(x);
                }
            }
        }
    }
    __syncthreads();

    // ---- stage 3: X3 = silu(X2 @ W3 + b3)   [64x128] -----------------------
    // 32 output tiles = 8 strip-units of (2 M-tiles x 2 n-tiles); 1 pass.
    {
        const __bf16* Bf = w3f + (size_t)t * (8 * 8) * 512;
        int mpair  = wv & 1;
        int ntile0 = (wv >> 1) * 2;
        v8f acc[2][2];
#pragma unroll
        for (int i = 0; i < 2; ++i)
#pragma unroll
            for (int j = 0; j < 2; ++j) acc[i][j] = (v8f){0,0,0,0,0,0,0,0};
        gemm_strip2<8, 2>(bufA, STR2, mpair * 32, Bf, 8, ntile0, lane, acc);
#pragma unroll
        for (int mi = 0; mi < 2; ++mi) {
            int mrow0 = mpair * 32 + mi * 16;
#pragma unroll
            for (int nt = 0; nt < 2; ++nt) {
                int n = (ntile0 + nt) * 16 + nlan;
                float bb = b3[t * 128 + n];
#pragma unroll
                for (int r = 0; r < 8; ++r) {
                    float x = acc[mi][nt][r] + bb;
                    bufB[(mrow0 + r + mlo) * STR3 + n] = (__bf16)siluf(x);
                }
            }
        }
    }
    __syncthreads();

    // ---- stage 4: edge_inv = tanh(X3 @ W4 + b4) * 10; scatter contrib ------
    {
        int pe = tid >> 6, eL = tid & 63;
        float s = 0.0f;
        const float* w4 = W4 + t * 128;
#pragma unroll
        for (int j = 0; j < 32; ++j)
            s += (float)bufB[eL * STR3 + pe * 32 + j] * w4[pe * 32 + j];
        part[eL * 4 + pe] = s;
    }
    __syncthreads();
    if (tid < 64) {
        float s = part[tid * 4] + part[tid * 4 + 1] + part[tid * 4 + 2] + part[tid * 4 + 3] + b4[t];
        float inv = tanhf(s) * CRANGE;
        int e = e0 + tid;
        if (emask[(size_t)t * EE + e]) {
            float el = elen[e];
            int rn = row_s[tid], cn = col_s[tid];
#pragma unroll
            for (int c = 0; c < 3; ++c) {
                float dd = (coord[rn * 3 + c] - coord[cn * 3 + c]) / el;
                float v  = dd * inv;
                atomicAdd(node_eq + ((size_t)t * NN + rn) * 3 + c,  v);
                atomicAdd(node_eq + ((size_t)t * NN + cn) * 3 + c, -v);
            }
        }
    }
}

// ---------------------------------------------------------------------------
// Node attention + combine: 16 nodes per block (64 rows = node x type), 4 waves.
// ---------------------------------------------------------------------------
#define WSTR  296   // stride for K=288 w tile
#define QSTR  136   // stride for 128-wide q/k/v
#define SN_W  0u
#define SN_Q  (64u * WSTR * 2u)                 // 37888
#define SN_K  (SN_Q + 64u * QSTR * 2u)
#define SN_V  (SN_K + 64u * QSTR * 2u)
#define SN_O  (SN_V + 64u * QSTR * 2u)          // f32 16*4*128
#define SM_NODE_TOTAL (SN_O + 16u * 4u * 128u * 4u)   // 122880

__global__ void __launch_bounds__(128, 1)
k_node_attn(const float* __restrict__ h,
            const float* __restrict__ coord,
            const float* __restrict__ cnt0,
            const float* __restrict__ cnt1,
            const float* __restrict__ sum0,
            const float* __restrict__ sum1,
            const __bf16* __restrict__ wqf,
            const __bf16* __restrict__ wkf,
            const __bf16* __restrict__ wvf,
            const float* __restrict__ Wo,
            const float* __restrict__ node_eq,
            float* __restrict__ out) {
    extern __shared__ char smem[];
    __bf16* wt  = (__bf16*)(smem + SN_W);
    __bf16* qb  = (__bf16*)(smem + SN_Q);
    __bf16* kb  = (__bf16*)(smem + SN_K);
    __bf16* vb  = (__bf16*)(smem + SN_V);
    float*  ob  = (float*)(smem + SN_O);

    const int tid  = threadIdx.x;                 // 128 threads
    const int lane = tid & 31;
    const int wv   = tid >> 5;                    // 0..3
    const int n0   = blockIdx.x * 16;
    const int mlo  = (lane & 16) ? 8 : 0;
    const int nlan = lane & 15;

    // ---- build w tile [64 rows x 288] bf16 ---------------------------------
    for (int it = tid; it < 64 * 64; it += 128) {
        int r = it >> 6, c = (it & 63) * 4;
        int nd = n0 + (r >> 2), ty = r & 3;
        v4f f = *(const v4f*)(h + ((size_t)nd * 4 + ty) * HH + c);
        __bf16* q = wt + r * WSTR + c;
        q[0] = (__bf16)f.x; q[1] = (__bf16)f.y; q[2] = (__bf16)f.z; q[3] = (__bf16)f.w;
    }
    if (tid < 64) {
        int r = tid, nd = n0 + (r >> 2), ty = r & 3;
        float m0 = sum0[ty * NN + nd] / fmaxf(cnt0[ty * NN + nd], 1.0f);
        float m1 = sum1[ty * NN + nd] / fmaxf(cnt1[ty * NN + nd], 1.0f);
        wt[r * WSTR + 256] = (__bf16)m0;
        wt[r * WSTR + 257] = (__bf16)m1;
        for (int j = DWK; j < WSTR; ++j) wt[r * WSTR + j] = (__bf16)0.0f;
    }
    __syncthreads();

    // ---- q/k/v projections via WMMA (K=288, N=128) -------------------------
    // 32 output tiles per matrix = 4 strip-units of (2 M-tiles x 4 n-tiles).
    const __bf16* wf[3]   = { wqf, wkf, wvf };
    __bf16*       outb[3] = { qb, kb, vb };
    const int mpair  = wv & 1;
    const int ntile0 = (wv >> 1) * 4;
#pragma unroll 1
    for (int mat = 0; mat < 3; ++mat) {
        v8f acc[2][4];
#pragma unroll
        for (int i = 0; i < 2; ++i)
#pragma unroll
            for (int j = 0; j < 4; ++j) acc[i][j] = (v8f){0,0,0,0,0,0,0,0};
        gemm_strip2<9, 4>(wt, WSTR, mpair * 32, wf[mat], 8, ntile0, lane, acc);
#pragma unroll
        for (int mi = 0; mi < 2; ++mi) {
            int mrow0 = mpair * 32 + mi * 16;
#pragma unroll
            for (int nt = 0; nt < 4; ++nt) {
                int n = (ntile0 + nt) * 16 + nlan;
#pragma unroll
                for (int r = 0; r < 8; ++r)
                    outb[mat][(mrow0 + r + mlo) * QSTR + n] = (__bf16)acc[mi][nt][r];
            }
        }
    }
    __syncthreads();

    // ---- tiny 4x4 attention per (node, head) -------------------------------
    if (tid < 64) {
        int head = tid & 3, nl = tid >> 2;
        int co = head * HDIM;
        float sc[4][4];
        for (int a = 0; a < 4; ++a)
            for (int s = 0; s < 4; ++s) {
                float acc = 0.0f;
#pragma unroll
                for (int d = 0; d < HDIM; ++d)
                    acc += (float)qb[(nl * 4 + a) * QSTR + co + d] *
                           (float)kb[(nl * 4 + s) * QSTR + co + d];
                sc[a][s] = acc * 0.17677669529663687f;   // 1/sqrt(32)
            }
        for (int a = 0; a < 4; ++a) {
            float mx = fmaxf(fmaxf(sc[a][0], sc[a][1]), fmaxf(sc[a][2], sc[a][3]));
            float su = 0.0f;
            for (int s = 0; s < 4; ++s) { sc[a][s] = __expf(sc[a][s] - mx); su += sc[a][s]; }
            float r = 1.0f / su;
            for (int s = 0; s < 4; ++s) sc[a][s] *= r;
        }
        for (int d = 0; d < HDIM; ++d) {
            float vv[4];
#pragma unroll
            for (int s = 0; s < 4; ++s)
                vv[s] = (float)vb[(nl * 4 + s) * QSTR + co + d];
#pragma unroll
            for (int a = 0; a < 4; ++a)
                ob[(nl * 4 + a) * AHD + co + d] =
                    sc[a][0] * vv[0] + sc[a][1] * vv[1] + sc[a][2] * vv[2] + sc[a][3] * vv[3];
        }
    }
    __syncthreads();

    // ---- type softmax + combine node_eq + coord ----------------------------
    if (tid < 16) {
        int nl = tid, nd = n0 + nl;
        float lg[4];
        for (int a = 0; a < 4; ++a) {
            float s = 0.0f;
            for (int j = 0; j < AHD; ++j) s += ob[(nl * 4 + a) * AHD + j] * Wo[j];
            lg[a] = s;
        }
        float mx = fmaxf(fmaxf(lg[0], lg[1]), fmaxf(lg[2], lg[3]));
        float su = 0.0f;
        for (int a = 0; a < 4; ++a) { lg[a] = __expf(lg[a] - mx); su += lg[a]; }
        float r = 1.0f / su;
#pragma unroll
        for (int c = 0; c < 3; ++c) {
            float p = 0.0f;
            for (int a = 0; a < 4; ++a)
                p += lg[a] * r * node_eq[((size_t)a * NN + nd) * 3 + c];
            out[nd * 3 + c]          = coord[nd * 3 + c] + p;   // coord_out
            out[NN * 3 + nd * 3 + c] = p;                       // node_eq_pos
        }
    }
}

// ---------------------------------------------------------------------------
// Host launch
// ---------------------------------------------------------------------------
extern "C" void kernel_launch(void* const* d_in, const int* in_sizes, int n_in,
                              void* d_out, int out_size, void* d_ws, size_t ws_size,
                              hipStream_t stream) {
    const float* h      = (const float*)d_in[0];
    const float* coord  = (const float*)d_in[1];
    const float* eattr  = (const float*)d_in[2];
    const float* elen   = (const float*)d_in[3];
    const int*   eidx   = (const int*)d_in[4];
    const unsigned char* emask = (const unsigned char*)d_in[5];
    const float* W1 = (const float*)d_in[6];
    const float* b1 = (const float*)d_in[7];
    const float* W2 = (const float*)d_in[8];
    const float* b2 = (const float*)d_in[9];
    const float* W3 = (const float*)d_in[10];
    const float* b3 = (const float*)d_in[11];
    const float* W4 = (const float*)d_in[12];
    const float* b4 = (const float*)d_in[13];
    const float* Wq = (const float*)d_in[14];
    const float* Wk = (const float*)d_in[15];
    const float* Wv = (const float*)d_in[16];
    const float* Wo = (const float*)d_in[17];
    float* out = (float*)d_out;

    char* ws = (char*)d_ws;
    float*  node_eq = (float*)(ws + OFF_NODE_EQ);
    float*  cnt0 = (float*)(ws + OFF_CNT0);
    float*  cnt1 = (float*)(ws + OFF_CNT1);
    float*  sum0 = (float*)(ws + OFF_SUM0);
    float*  sum1 = (float*)(ws + OFF_SUM1);
    __bf16* w1f  = (__bf16*)(ws + OFF_W1FRAG);
    float*  w1l  = (float*)(ws + OFF_W1LAST);
    __bf16* w2f  = (__bf16*)(ws + OFF_W2FRAG);
    __bf16* w3f  = (__bf16*)(ws + OFF_W3FRAG);
    __bf16* wqf  = (__bf16*)(ws + OFF_WQFRAG);
    __bf16* wkf  = (__bf16*)(ws + OFF_WKFRAG);
    __bf16* wvf  = (__bf16*)(ws + OFF_WVFRAG);

    // 0) zero accumulators
    k_zero<<<(ZERO_FLOATS + 255) / 256, 256, 0, stream>>>(node_eq, ZERO_FLOATS);

    // 1) weight prep -> WMMA fragment layouts (bf16)
    k_fragconv<<<256, 256, 0, stream>>>(W1, w1f, 512, 512, 16, 32, 4, (long)513 * 512);
    k_w1last<<<8, 256, 0, stream>>>(W1, w1l);
    k_fragconv<<<128, 256, 0, stream>>>(W2, w2f, 512, 256, 16, 16, 4, (long)512 * 256);
    k_fragconv<<<32, 256, 0, stream>>>(W3, w3f, 256, 128, 8, 8, 4, (long)256 * 128);
    k_fragconv<<<9, 256, 0, stream>>>(Wq, wqf, DWK, AHD, 9, 8, 1, 0);
    k_fragconv<<<9, 256, 0, stream>>>(Wk, wkf, DWK, AHD, 9, 8, 1, 0);
    k_fragconv<<<9, 256, 0, stream>>>(Wv, wvf, DWK, AHD, 9, 8, 1, 0);

    // 2) masked per-node edge statistics
    k_edge_stats<<<EE / 256, 256, 0, stream>>>(eidx, emask, eattr, cnt0, cnt1, sum0, sum1);

    // 3) fused edge MLP + coordinate scatter  (grid: 1024 edge-tiles x 4 types)
    dim3 gmlp(EE / 64, 4);
    k_edge_mlp<<<gmlp, 256, SM_EDGE_TOTAL, stream>>>(
        h, coord, eattr, elen, eidx, emask,
        w1f, w1l, b1, w2f, b2, w3f, b3, W4, b4, node_eq);

    // 4) node attention + final combine
    k_node_attn<<<NN / 16, 128, SM_NODE_TOTAL, stream>>>(
        h, coord, cnt0, cnt1, sum0, sum1, wqf, wkf, wvf, Wo, node_eq, out);
}
